// kgpr_46316927320314
// MI455X (gfx1250) — matrix operand order repaired and checked by reading, not compiled, not run on hardware
//
#include <hip/hip_runtime.h>

// ---------------------------------------------------------------------------
// RBF-kernel Nystrom head:  y[b, h*64+d] = sum_m exp(-beta*||z_b - R_m||^2) * M_hX[h,m,d]
// B=16384, d_lat=64, M=2048, H=4, D=64.  ~21.5 GFLOP over ~24MB -> compute bound.
// Both GEMMs mapped to v_wmma_f32_16x16x32_bf16 (deepest-K 16-bit WMMA on CDNA5).
// ---------------------------------------------------------------------------

typedef __attribute__((ext_vector_type(16))) __bf16 v16bf;
typedef __attribute__((ext_vector_type(8)))  float  v8f;

#define BETA   0.05f
#define Bn     16384
#define Mn     2048
#define DLAT   64
#define NCOL   256            // H*D
#define BROWS  128            // B-rows per block (8 waves x 16 rows)
#define MT     32             // m-chunk (== WMMA K)
#define NCHUNK (Mn / MT)      // 64
#define NTILE  (NCOL / 16)    // 16 column tiles per wave

// workspace layout (bytes):
//   R2  : float [2048]            @ 0        (8 KB)
//   Rbf : __bf16[2048*64]         @ 8192     (256 KB)
//   W   : __bf16[64*16*32*16]     @ 270336   (1 MB)  pre-swizzled B2 fragments

// --- prep: R -> bf16 row-major + R2 = ||R_m||^2 ------------------------------
__global__ void prep_R(const float* __restrict__ R, float* __restrict__ R2,
                       __bf16* __restrict__ Rbf) {
  int m = blockIdx.x * blockDim.x + threadIdx.x;
  if (m >= Mn) return;
  float acc = 0.f;
#pragma unroll 8
  for (int k = 0; k < DLAT; ++k) {
    float r = R[m * DLAT + k];
    acc += r * r;
    Rbf[m * DLAT + k] = (__bf16)r;
  }
  R2[m] = acc;
}

// --- prep: M_hX -> per-(chunk, col-tile) B-fragment layout -------------------
// W[((c*16 + t)*32 + lane)*16 + i] = bf16( M_hX[h, m0 + i + 16*(lane>>4), d] )
// with h = t>>2, d = (t&3)*16 + (lane&15)  ->  hot-loop B loads are 2x b128/lane.
__global__ void prep_W(const float* __restrict__ Mx, __bf16* __restrict__ W) {
  int idx = blockIdx.x * blockDim.x + threadIdx.x;   // 524288 total
  int c = idx >> 13;
  int t = (idx >> 9) & 15;
  int l = (idx >> 4) & 31;
  int i = idx & 15;
  int m = c * MT + i + ((l >> 4) << 4);
  int h = t >> 2;
  int d = ((t & 3) << 4) + (l & 15);
  W[idx] = (__bf16)Mx[(h * Mn + m) * DLAT + d];
}

// --- main fused kernel -------------------------------------------------------
__launch_bounds__(256, 1)
__global__ void rbf_nystrom(const float* __restrict__ z, const float* __restrict__ R2,
                            const __bf16* __restrict__ Rbf, const __bf16* __restrict__ W,
                            float* __restrict__ out) {
  __shared__ float  shx2[8][16];
  __shared__ __bf16 kst[8][16][MT];     // per-wave k-tile staging (16 rows x 32 m)

  const int wave    = threadIdx.x >> 5;
  const int lane    = threadIdx.x & 31;
  const int lhalf   = lane >> 4;        // 0 / 1
  const int l15     = lane & 15;
  const int rowbase = blockIdx.x * BROWS + wave * 16;

  // A1 fragments: z rows in bf16; K=64 split into two K=32 frags.
  // A layout: lane (l%16)=row; half i -> k = i + 16*(l>>4): contiguous.
  v16bf a1[2];
  {
    const float* zr = z + (size_t)(rowbase + l15) * DLAT;
#pragma unroll
    for (int f = 0; f < 2; ++f)
#pragma unroll
      for (int i = 0; i < 16; ++i) {
        int k = f * 32 + lhalf * 16 + i;
        a1[f][i] = (__bf16)zr[k];
      }
  }

  // x2 = ||z_row||^2 (one lane per row), shared so any lane can pick its C-row's x2.
  if (lane < 16) {
    const float* zr = z + (size_t)(rowbase + lane) * DLAT;
    float acc = 0.f;
#pragma unroll 8
    for (int k = 0; k < DLAT; ++k) acc += zr[k] * zr[k];
    shx2[wave][lane] = acc;
  }
  __syncthreads();   // once per block: x2 exchange (cross-wave-safe, cheap)
  float x2v[8];
#pragma unroll
  for (int v = 0; v < 8; ++v) x2v[v] = shx2[wave][v + lhalf * 8];

  v8f acc[NTILE];
  {
    v8f zero = {};
#pragma unroll
    for (int t = 0; t < NTILE; ++t) acc[t] = zero;
  }

  for (int c = 0; c < NCHUNK; ++c) {
    const int m0 = c * MT;

    // ---- phase A: k-tile = exp(-beta * max(x2 + R2 - 2*z.R, 0)), 16x32 ----
#pragma unroll
    for (int nt = 0; nt < 2; ++nt) {
      const int m = m0 + nt * 16 + l15;
      const __bf16* rrow = Rbf + (size_t)m * DLAT;
      // B layout: lane (l%16)=col(m); half i -> k_zdim = i + 16*(l>>4): contiguous.
      v16bf b1lo = *(const v16bf*)(rrow + lhalf * 16);        // z-dims 0..31
      v16bf b1hi = *(const v16bf*)(rrow + 32 + lhalf * 16);   // z-dims 32..63
      v8f c1 = {};
      c1 = __builtin_amdgcn_wmma_f32_16x16x32_bf16(false, a1[0], false, b1lo,
                                                   (short)0, c1, false, false);
      c1 = __builtin_amdgcn_wmma_f32_16x16x32_bf16(false, a1[1], false, b1hi,
                                                   (short)0, c1, false, false);
      const float r2 = R2[m];
#pragma unroll
      for (int v = 0; v < 8; ++v) {
        // C layout: row = v + 8*(l>>4), col = l%16
        float d2 = x2v[v] + r2 - 2.0f * c1[v];
        d2 = fmaxf(d2, 0.f);
        kst[wave][v + lhalf * 8][nt * 16 + l15] = (__bf16)__expf(-BETA * d2);
      }
    }
    // kst region is wave-private and same-wave DS ops are in-order in HW;
    // only a compiler scheduling fence is needed (emits no instructions).
    __builtin_amdgcn_wave_barrier();

    // ---- repackage k-tile as A2 fragment (C layout != A layout -> via LDS) ----
    union { struct { uint4 lo, hi; } u; v16bf v; } a2u;
    const __bf16* ksrow = &kst[wave][l15][0];
    a2u.u.lo = *(const uint4*)(ksrow + lhalf * 16);       // halves 0..7
    a2u.u.hi = *(const uint4*)(ksrow + lhalf * 16 + 8);   // halves 8..15
    const v16bf a2 = a2u.v;
    __builtin_amdgcn_wave_barrier();   // WAR fence vs next chunk's stores

    // ---- phase B: acc[t] += k_tile @ M_hX_tile (pre-swizzled fragments) ----
    const __bf16* wbase = W + (size_t)c * (NTILE * 512) + lane * 16;
#pragma unroll
    for (int t = 0; t < NTILE; ++t) {
      v16bf wb = *(const v16bf*)(wbase + t * 512);
      acc[t] = __builtin_amdgcn_wmma_f32_16x16x32_bf16(false, a2, false, wb,
                                                       (short)0, acc[t], false, false);
    }
  }

  // ---- writeback: C layout row = v + 8*(l>>4), col = 16*t + l%16 ----
  // Lanes 0-15 / 16-31 each cover 64B contiguous spans -> fully coalesced b32s.
#pragma unroll
  for (int t = 0; t < NTILE; ++t)
#pragma unroll
    for (int v = 0; v < 8; ++v)
      out[(size_t)(rowbase + v + lhalf * 8) * NCOL + t * 16 + l15] = acc[t][v];
}

extern "C" void kernel_launch(void* const* d_in, const int* in_sizes, int n_in,
                              void* d_out, int out_size, void* d_ws, size_t ws_size,
                              hipStream_t stream) {
  (void)in_sizes; (void)n_in; (void)out_size; (void)ws_size;
  const float* z  = (const float*)d_in[0];
  const float* R  = (const float*)d_in[1];
  const float* Mx = (const float*)d_in[2];
  float* out = (float*)d_out;

  float*  R2w = (float*)d_ws;
  __bf16* Rbf = (__bf16*)((char*)d_ws + 8192);
  __bf16* Wp  = (__bf16*)((char*)d_ws + 270336);

  prep_R<<<(Mn + 255) / 256, 256, 0, stream>>>(R, R2w, Rbf);
  prep_W<<<(NCHUNK * NTILE * 32 * 16) / 256, 256, 0, stream>>>(Mx, Wp);
  rbf_nystrom<<<Bn / BROWS, 256, 0, stream>>>(z, R2w, Rbf, Wp, out);
}
